// NeuralNetwork_33646773796865
// MI455X (gfx1250) — 8-bit matrix operand order repaired and checked by reading, compiled, not run on hardware
//
#include <hip/hip_runtime.h>

typedef __attribute__((ext_vector_type(8))) int v8i;
typedef __attribute__((ext_vector_type(4))) int v4i;
typedef unsigned long long u64;

#define MBATCH 16384
#define GEMM_MT 256
#define GEMM_NT 64

__global__ void binarize_act_kernel(const float* __restrict__ in,
                                    signed char* __restrict__ out,
                                    int cols, int colsPad) {
    size_t idx = (size_t)blockIdx.x * 256 + threadIdx.x;
    int n = (int)(idx % colsPad);
    size_t m = idx / colsPad;
    signed char r = 0;
    if (n < cols) r = (in[m * cols + n] >= 0.0f) ? (signed char)1 : (signed char)-1;
    out[idx] = r;
}

__global__ void binarize_w_kernel(const float* __restrict__ W,
                                  signed char* __restrict__ out,
                                  int fout, int fin, int Kp) {
    size_t idx = (size_t)blockIdx.x * 256 + threadIdx.x;
    int k = (int)(idx % Kp);
    int n = (int)(idx / Kp);
    signed char r = 0;
    if (n < fout && k < fin)
        r = (W[(size_t)n * fin + k] >= 0.0f) ? (signed char)1 : (signed char)-1;
    out[idx] = r;
}

__launch_bounds__(256)
__global__ void gemm_bin_kernel(const signed char* __restrict__ A,
                                const signed char* __restrict__ Bw,
                                int* __restrict__ C, int Kp, int Np) {
    __shared__ signed char As[2][GEMM_MT * 64];
    __shared__ signed char Bs[2][GEMM_NT * 64];

    const int tid  = threadIdx.x;
    const int lane = tid & 31;
    const int wave = tid >> 5;
    const int blockM = blockIdx.x * GEMM_MT;
    const int blockN = blockIdx.y * GEMM_NT;

    const int sRow = tid >> 2;
    const int sCh  = (tid & 3) * 16;
    const signed char* gA = &A[(size_t)(blockM + sRow) * Kp + sCh];
    const signed char* gB = &Bw[(size_t)(blockN + sRow) * Kp + sCh];
    const size_t aRowStep = (size_t)64 * Kp;

    v8i acc[2][4];
#pragma unroll
    for (int i = 0; i < 2; ++i)
#pragma unroll
        for (int j = 0; j < 4; ++j)
#pragma unroll
            for (int r = 0; r < 8; ++r) acc[i][j][r] = 0;

    const int nIter = Kp / 64;

    auto issue = [&](int it, int buf) {
        const int k0 = it * 64;
        unsigned ldsA = (unsigned)(size_t)&As[buf][sRow * 64 + sCh];
        unsigned ldsB = (unsigned)(size_t)&Bs[buf][sRow * 64 + sCh];
#pragma unroll
        for (int j = 0; j < 4; ++j) {
            asm volatile("global_load_async_to_lds_b128 %0, %1, off"
                         :: "v"(ldsA + (unsigned)(j * 64 * 64)),
                            "v"(gA + (size_t)j * aRowStep + k0)
                         : "memory");
        }
        asm volatile("global_load_async_to_lds_b128 %0, %1, off"
                     :: "v"(ldsB), "v"(gB + k0) : "memory");
    };

    issue(0, 0);

    for (int it = 0; it < nIter; ++it) {
        const int buf = it & 1;
        if (it + 1 < nIter) {
            issue(it + 1, buf ^ 1);
            asm volatile("s_wait_asynccnt 0x5" ::: "memory");
        } else {
            asm volatile("s_wait_asynccnt 0x0" ::: "memory");
        }
        __syncthreads();

        const int kbase = (lane >> 4) * 16;
        const int arow  = wave * 32 + (lane & 15);
        v8i afrag[2];
#pragma unroll
        for (int i = 0; i < 2; ++i) {
            const signed char* p = &As[buf][(arow + i * 16) * 64 + kbase];
            union { v8i v; u64 q[4]; } u;
            u.q[0] = *(const u64*)(p);
            u.q[1] = *(const u64*)(p + 8);
            u.q[2] = *(const u64*)(p + 32);
            u.q[3] = *(const u64*)(p + 40);
            afrag[i] = u.v;
        }
        const int bn = lane & 15;
        const int kh = (lane >> 4) * 16;
        v8i bfrag[4];
#pragma unroll
        for (int j = 0; j < 4; ++j) {
            const signed char* p = &Bs[buf][(j * 16 + bn) * 64 + kh];
            union { v8i v; v4i h[2]; } u;
            u.h[0] = *(const v4i*)(p);
            u.h[1] = *(const v4i*)(p + 32);
            bfrag[j] = u.v;
        }

#pragma unroll
        for (int i = 0; i < 2; ++i)
#pragma unroll
            for (int j = 0; j < 4; ++j)
                acc[i][j] = __builtin_amdgcn_wmma_i32_16x16x64_iu8(
                     true, afrag[i],  true, bfrag[j],
                    acc[i][j],  false,  false);
        __syncthreads();
    }

    const int crow0 = blockM + wave * 32 + ((lane >> 4) * 8);
    const int ccol0 = blockN + (lane & 15);
#pragma unroll
    for (int i = 0; i < 2; ++i)
#pragma unroll
        for (int j = 0; j < 4; ++j)
#pragma unroll
            for (int r = 0; r < 8; ++r)
                C[(size_t)(crow0 + i * 16 + r) * Np + (ccol0 + j * 16)] = acc[i][j][r];
}

__global__ void colstats_kernel(const int* __restrict__ C,
                                float* __restrict__ meanv,
                                float* __restrict__ rstdv, int Np) {
    __shared__ float s1[256];
    __shared__ float s2[256];
    const int n = blockIdx.x;
    const int tid = threadIdx.x;
    float sum = 0.0f, sq = 0.0f;
    for (int m = tid; m < MBATCH; m += 256) {
        float v = (float)C[(size_t)m * Np + n];
        sum += v;
        sq += v * v;
    }
    s1[tid] = sum;
    s2[tid] = sq;
    __syncthreads();
    for (int s = 128; s > 0; s >>= 1) {
        if (tid < s) { s1[tid] += s1[tid + s]; s2[tid] += s2[tid + s]; }
        __syncthreads();
    }
    if (tid == 0) {
        float mean = s1[0] * (1.0f / MBATCH);
        float var  = s2[0] * (1.0f / MBATCH) - mean * mean;
        meanv[n] = mean;
        rstdv[n] = rsqrtf(var + 1e-5f);
    }
}

__global__ void threshold_kernel(const int* __restrict__ C,
                                 const float* __restrict__ meanv,
                                 const float* __restrict__ rstdv,
                                 const float* __restrict__ g,
                                 const float* __restrict__ be,
                                 signed char* __restrict__ out,
                                 int fout, int Np) {
    size_t idx = (size_t)blockIdx.x * 256 + threadIdx.x;
    int n = (int)(idx % Np);
    signed char r = 0;
    if (n < fout) {
        float v = ((float)C[idx] - meanv[n]) * rstdv[n] * g[n] + be[n];
        r = (v >= 0.0f) ? (signed char)1 : (signed char)-1;
    }
    out[idx] = r;
}

__global__ void final_out_kernel(const int* __restrict__ C,
                                 const float* __restrict__ b6,
                                 float* __restrict__ out, int Np) {
    int idx = blockIdx.x * 256 + threadIdx.x;
    if (idx >= MBATCH * 35) return;
    int m = idx / 35, n = idx % 35;
    float sb = (b6[n] >= 0.0f) ? 1.0f : -1.0f;
    out[idx] = (float)C[(size_t)m * Np + n] + sb;
}

extern "C" void kernel_launch(void* const* d_in, const int* in_sizes, int n_in,
                              void* d_out, int out_size, void* d_ws, size_t ws_size,
                              hipStream_t stream) {
    (void)in_sizes; (void)n_in; (void)out_size; (void)ws_size;

    static const int FIN[6]  = {4096, 500, 400, 350, 300, 300};
    static const int FOUT[6] = {500, 400, 350, 300, 300, 35};
    static const int KP[6]   = {4096, 512, 448, 384, 320, 320};
    static const int NP[6]   = {512, 448, 384, 320, 320, 64};

    const float* x = (const float*)d_in[0];

    char* ws = (char*)d_ws;
    signed char* actX = (signed char*)ws;
    signed char* buf1 = actX + (size_t)MBATCH * 4096;
    signed char* buf2 = buf1 + (size_t)MBATCH * 512;
    int* accum = (int*)(buf2 + (size_t)MBATCH * 512);
    signed char* wbin = (signed char*)(accum + (size_t)MBATCH * 512);
    float* meanv = (float*)(wbin + (size_t)512 * 4096);
    float* rstdv = meanv + 512;

    binarize_act_kernel<<<(MBATCH * 4096) / 256, 256, 0, stream>>>(x, actX, 4096, 4096);

    const signed char* curAct = actX;
    signed char* bufs[2] = {buf1, buf2};

    for (int L = 0; L < 6; ++L) {
        const int fin = FIN[L], fout = FOUT[L], Kp = KP[L], Np = NP[L];
        const float* W = (const float*)d_in[1 + L * 4];
        const float* b = (const float*)d_in[2 + L * 4];

        binarize_w_kernel<<<(Np * Kp) / 256, 256, 0, stream>>>(W, wbin, fout, fin, Kp);

        dim3 grid(MBATCH / GEMM_MT, Np / GEMM_NT);
        gemm_bin_kernel<<<grid, 256, 0, stream>>>(curAct, wbin, accum, Kp, Np);

        if (L < 5) {
            const float* g  = (const float*)d_in[3 + L * 4];
            const float* be = (const float*)d_in[4 + L * 4];
            colstats_kernel<<<Np, 256, 0, stream>>>(accum, meanv, rstdv, Np);
            threshold_kernel<<<(MBATCH * Np) / 256, 256, 0, stream>>>(
                accum, meanv, rstdv, g, be, bufs[L & 1], fout, Np);
            curAct = bufs[L & 1];
        } else {
            final_out_kernel<<<(MBATCH * 35 + 255) / 256, 256, 0, stream>>>(
                accum, b, (float*)d_out, Np);
        }
    }
}
